// TernaryConv2d_42597485641963
// MI455X (gfx1250) — matrix operand order repaired and checked by reading, not compiled, not run on hardware
//
#include <hip/hip_runtime.h>

// ---------------------------------------------------------------------------
// TWN ternary 3x3 conv as implicit GEMM on CDNA5 WMMA (gfx1250, wave32).
//   M = 256 out-channels, N = 93312 output pixels, K = 1152 (= 128*3*3)
//   Workgroup tile: 256(M) x 32(N); 8 waves, each wave = 4 M-tiles x 1 N-tile
//   A (16x32 f16)  = ternary weights (exact in f16), pre-packed fragment order
//   B (32x16 f16)  = activations, f32 split hi/lo f16 (2 WMMAs per K-chunk)
//   C/D (16x16 f32) accumulate in f32 -> matches f32 reference ~1e-6.
//   Software pipeline: LDS ping-pong for B (1 barrier/chunk) + register
//   double-buffer for A. Each thread stages 4 CONSECUTIVE slots (one pixel,
//   4 consecutive K): 1x ds_load_b128 (offset table) + 2x ds_store_b64.
// ---------------------------------------------------------------------------

typedef __attribute__((ext_vector_type(16))) _Float16 v16h;
typedef __attribute__((ext_vector_type(4)))  _Float16 v4h;
typedef __attribute__((ext_vector_type(8)))  float    v8f;
typedef __attribute__((ext_vector_type(4)))  int      v4i;

#define C_IN     128
#define C_OUT    256
#define HW_IN    56
#define HW_OUT   54
#define NIMG     32
#define KTOT     1152                  // 128*9
#define KCHUNKS  36                    // 1152/32
#define PIX_PER_IMG (HW_OUT*HW_OUT)    // 2916
#define NPIX     (NIMG*PIX_PER_IMG)    // 93312
#define NBLK     32                    // pixels per workgroup (2 N-tiles)
#define NWG      (NPIX/NBLK)           // 2916 workgroups
#define IN_PLANE (HW_IN*HW_IN)         // 3136
#define WGS      256
#define WS_A_OFF 256                   // bytes; ws[0] = threshold

#define WMMA_F16(a, b, c) \
  __builtin_amdgcn_wmma_f32_16x16x32_f16(false, (a), false, (b), (short)0, (c), false, false)

// ---------------- kernel 1: threshold = 0.05 * max|w| (one workgroup) -------
__global__ __launch_bounds__(256) void twn_threshold_kernel(
    const float* __restrict__ w, float* __restrict__ thr) {
  __shared__ float red[256];
  float m = 0.0f;
  for (int i = threadIdx.x; i < C_OUT * KTOT; i += 256)
    m = fmaxf(m, fabsf(w[i]));
  red[threadIdx.x] = m;
  __syncthreads();
  for (int s = 128; s > 0; s >>= 1) {
    if ((int)threadIdx.x < s)
      red[threadIdx.x] = fmaxf(red[threadIdx.x], red[threadIdx.x + s]);
    __syncthreads();
  }
  if (threadIdx.x == 0) thr[0] = 0.05f * red[0];
}

// ---------------- kernel 2: quantize + pack into A-fragment layout ----------
// A fragment (16-bit A 16x32, ISA 7.12.2): lane = m + 16*half,
//   element e in v16h:  K = (e<8 ? e : e+8) + 8*half
// inverse: half = (kk>>3)&1 ; e = (kk&7) + ((kk&16)?8:0)
__global__ __launch_bounds__(256) void twn_quantize_pack_kernel(
    const float* __restrict__ w, const float* __restrict__ thr,
    _Float16* __restrict__ afrag) {
  int idx = blockIdx.x * 256 + threadIdx.x;      // idx = o*1152 + k (OIHW flat)
  float t  = thr[0];
  float wv = w[idx];
  float q  = (wv > t ? 1.0f : 0.0f) - (wv < -t ? 1.0f : 0.0f);
  int o = idx / KTOT, k = idx - o * KTOT;
  int mtile = o >> 4, m = o & 15;
  int kchunk = k >> 5, kk = k & 31;
  int half = (kk >> 3) & 1;
  int e    = (kk & 7) + ((kk & 16) ? 8 : 0);
  int lane = m + (half << 4);
  afrag[((((size_t)mtile * KCHUNKS + kchunk) * 32 + lane) << 4) + e] = (_Float16)q;
}

// ---------------- kernel 3: implicit-GEMM conv with WMMA --------------------
__global__ __launch_bounds__(256) void twn_conv_wmma_kernel(
    const float* __restrict__ x, const _Float16* __restrict__ afrag,
    const float* __restrict__ bias, float* __restrict__ out) {
  __shared__ int xoff_s[KTOT];                          // 4.6 KB: k -> x offset
  __shared__ __align__(32) _Float16 bhi_s[2 * 1024];    // [buf][ntile][blane][e]
  __shared__ __align__(32) _Float16 blo_s[2 * 1024];

  const int tid   = threadIdx.x;
  const int wave  = tid >> 5;
  const int lane  = tid & 31;
  const int ntile = wave >> 2;           // 0..1 : which 16-pixel column tile
  const int mq    = wave & 3;            // 0..3 : M-tiles 4*mq .. 4*mq+3
  const int pbase = blockIdx.x * NBLK;

  // Build gather-offset table once: off(k) = c*3136 + kh*56 + kw
  for (int k = tid; k < KTOT; k += WGS) {
    int c = k / 9, r = k - c * 9, kh = r / 3, kw = r - kh * 3;
    xoff_s[k] = c * IN_PLANE + kh * HW_IN + kw;
  }

  // Staging: thread owns 4 CONSECUTIVE slots f0..f0+3 (same blane => one pixel,
  // 4 consecutive K values). slot f: nt=f>>9, blane=(f&511)>>4, e=f&15,
  // K-in-chunk = (e&7) + 8*(blane>>4) + 16*(e>>3), pixel col = blane&15.
  const int f0    = 4 * tid;             // 0..1020, covers all 1024 slots
  const int blane = (f0 & 511) >> 4;
  const int kbin  = (f0 & 7) + ((blane >> 4) << 3) + ((f0 & 8) << 1);   // K base, j adds 0..3
  int pix;
  {
    int p  = pbase + (f0 >> 9) * 16 + (blane & 15);
    int ni = p / PIX_PER_IMG, rem = p - ni * PIX_PER_IMG;
    int oh = rem / HW_OUT,    ow  = rem - oh * HW_OUT;
    pix = ni * C_IN * IN_PLANE + oh * HW_IN + ow;
  }
  __syncthreads();   // xoff table ready

  const _Float16* ap[4];
#pragma unroll
  for (int t = 0; t < 4; ++t)
    ap[t] = afrag + (((size_t)(4 * mq + t) * KCHUNKS) * 32 + lane) * 16;

  v8f acc[4] = {{}, {}, {}, {}};

  // ---- pipeline prologue: stage chunk 0 into buf 0, preload A chunk 0 ----
  v16h aCur[4], aNext[4];
#pragma unroll
  for (int t = 0; t < 4; ++t)
    aCur[t] = *(const v16h*)(ap[t]);
  {
    v4i offs = *(const v4i*)(xoff_s + kbin);
    v4h h, l;
#pragma unroll
    for (int j = 0; j < 4; ++j) {
      float xv = x[pix + offs[j]];
      h[j] = (_Float16)xv;
      l[j] = (_Float16)(xv - (float)h[j]);
    }
    *(v4h*)(bhi_s + f0) = h;
    *(v4h*)(blo_s + f0) = l;
  }
  __syncthreads();   // buf 0 staged

  // ---- main pipelined loop: one barrier per K-chunk ----
  for (int chunk = 0; chunk < KCHUNKS; ++chunk) {
    const int buf   = chunk & 1;
    const bool more = (chunk + 1 < KCHUNKS);

    // issue next chunk's gathers + A loads; latency hidden under WMMAs below
    float xn[4];
    if (more) {
      v4i offs = *(const v4i*)(xoff_s + (chunk + 1) * 32 + kbin);
#pragma unroll
      for (int j = 0; j < 4; ++j)
        xn[j] = x[pix + offs[j]];
#pragma unroll
      for (int t = 0; t < 4; ++t)
        aNext[t] = *(const v16h*)(ap[t] + (size_t)(chunk + 1) * 512);
    }

    v16h bh = *(const v16h*)(bhi_s + buf * 1024 + ntile * 512 + lane * 16);
    v16h bl = *(const v16h*)(blo_s + buf * 1024 + ntile * 512 + lane * 16);
#pragma unroll
    for (int t = 0; t < 4; ++t) {
      acc[t] = WMMA_F16(aCur[t], bh, acc[t]);
      acc[t] = WMMA_F16(aCur[t], bl, acc[t]);
    }

    if (more) {
      const int nbuf = buf ^ 1;
      v4h h, l;
#pragma unroll
      for (int j = 0; j < 4; ++j) {
        h[j] = (_Float16)xn[j];
        l[j] = (_Float16)(xn[j] - (float)h[j]);
      }
      *(v4h*)(bhi_s + nbuf * 1024 + f0) = h;
      *(v4h*)(blo_s + nbuf * 1024 + f0) = l;
#pragma unroll
      for (int t = 0; t < 4; ++t)
        aCur[t] = aNext[t];
    }
    __syncthreads();   // publish buf^1 stores; fence this chunk's buf reads
  }

  // Epilogue: C/D 16x16 f32 layout -> col = lane&15, row = i + 8*(lane>>4)
  int col = lane & 15, rbase = (lane >> 4) * 8;
  int p  = pbase + ntile * 16 + col;
  int ni = p / PIX_PER_IMG, rr = p - ni * PIX_PER_IMG;
  int oh = rr / HW_OUT,     ow = rr - oh * HW_OUT;
  size_t obase = (size_t)ni * C_OUT * PIX_PER_IMG + (size_t)oh * HW_OUT + ow;
#pragma unroll
  for (int t = 0; t < 4; ++t) {
#pragma unroll
    for (int i = 0; i < 8; ++i) {
      int o = (4 * mq + t) * 16 + rbase + i;
      out[obase + (size_t)o * PIX_PER_IMG] = acc[t][i] + bias[o];
    }
  }
}

// ---------------------------------------------------------------------------
extern "C" void kernel_launch(void* const* d_in, const int* in_sizes, int n_in,
                              void* d_out, int out_size, void* d_ws, size_t ws_size,
                              hipStream_t stream) {
  const float* x    = (const float*)d_in[0];
  const float* w    = (const float*)d_in[1];
  const float* bias = (const float*)d_in[2];
  float* out = (float*)d_out;

  float*    thr   = (float*)d_ws;
  _Float16* afrag = (_Float16*)((char*)d_ws + WS_A_OFF);  // 16*36*32*16 f16 = 576 KB

  twn_threshold_kernel<<<1, 256, 0, stream>>>(w, thr);
  twn_quantize_pack_kernel<<<(C_OUT * KTOT) / 256, 256, 0, stream>>>(w, thr, afrag);
  twn_conv_wmma_kernel<<<NWG, 256, 0, stream>>>(x, afrag, bias, out);
}
